// FastMultiHeadAttention_23493471109255
// MI455X (gfx1250) — compile-verified, matrix operand order checked by reading, not run on hardware
//
#include <hip/hip_runtime.h>

typedef __bf16 bf16_t;
typedef __attribute__((ext_vector_type(16))) __bf16 v16bf;
typedef __attribute__((ext_vector_type(8)))  __bf16 v8bf;
typedef __attribute__((ext_vector_type(8)))  float  v8f;
typedef __attribute__((ext_vector_type(4)))  int    tdm4i;
typedef __attribute__((ext_vector_type(8)))  int    tdm8i;

#define DIM_  1024
#define NH_   16
#define HD_   64
#define B_    2
#define T_    2048
#define BH_   (B_ * NH_)   // 32

#define WMMA_BF16(A, Bv, C) \
  __builtin_amdgcn_wmma_f32_16x16x32_bf16(false, (A), false, (Bv), (short)0, (C), false, false)

static __device__ __forceinline__ unsigned short f2bf_u(float f) {
  unsigned u = __builtin_bit_cast(unsigned, f);
  u += 0x7FFFu + ((u >> 16) & 1u);          // round-to-nearest-even
  return (unsigned short)(u >> 16);
}
static __device__ __forceinline__ float bf2f(unsigned short h) {
  unsigned u = ((unsigned)h) << 16;
  return __builtin_bit_cast(float, u);
}

// Low 32 bits of a generic LDS address are the LDS byte offset (ISA 10.2:
// LDS aperture occupies addr[63:32]).
static __device__ __forceinline__ unsigned lds_offset(const void* p) {
  return (unsigned)(uintptr_t)p;
}

// ---------------------------------------------------------------------------
// Tensor Data Mover: 2D tile load Global->LDS, bf16 elements.
// D# per ISA cdna5 §8.3/8.4: group0 = {count|lds_addr|global_addr|type=2},
// group1 = {data_size=2B, tensor_dim0/1, tile_dim0/1, tensor_dim0_stride}.
// 2-group form (VADDR2/3 = NULL) covers 2D tensors. Tracked by TENSORcnt.
// ---------------------------------------------------------------------------
static __device__ __forceinline__ void tdm_load_tile_bf16(
    unsigned lds_off, const unsigned short* gptr,
    unsigned tensor_d0, unsigned tensor_d1,
    unsigned tile_d0, unsigned tile_d1, unsigned stride_elems) {
  const unsigned long long ga = (unsigned long long)(uintptr_t)gptr;
  tdm4i g0;
  g0[0] = 1;                                                   // count=1 (user)
  g0[1] = (int)lds_off;                                        // lds_addr
  g0[2] = (int)(unsigned)(ga & 0xFFFFFFFFull);                 // global_addr lo
  g0[3] = (int)(((unsigned)(ga >> 32) & 0x01FFFFFFu)           // global_addr hi
                | 0x80000000u);                                // type=2 ("image")
  tdm8i g1;
  g1[0] = 0x00010000;                                          // data_size=1 (2B)
  g1[1] = (int)(tensor_d0 << 16);                              // tensor_dim0 lo16
  g1[2] = (int)((tensor_d0 >> 16) | (tensor_d1 << 16));        // dim0 hi | dim1 lo
  g1[3] = (int)((tensor_d1 >> 16) | (tile_d0 << 16));          // dim1 hi | tile_dim0
  g1[4] = (int)(tile_d1 & 0xFFFFu);                            // tile_dim1 (dim2=0)
  g1[5] = (int)stride_elems;                                   // dim0_stride lo32
  g1[6] = 0;                                                   // stride hi16 | dim1_stride lo
  g1[7] = 0;
  asm volatile("tensor_load_to_lds %0, %1" :: "s"(g0), "s"(g1) : "memory");
}

// ---------------------------------------------------------------------------
// Kernel 1: fp32 -> bf16 cast (4 elements / thread)
// ---------------------------------------------------------------------------
__global__ __launch_bounds__(256) void cast_f32_to_bf16(
    const float* __restrict__ in, unsigned short* __restrict__ out, int n4) {
  int i = blockIdx.x * 256 + threadIdx.x;
  if (i >= n4) return;
  float4 v = reinterpret_cast<const float4*>(in)[i];
  ushort4 o;
  o.x = f2bf_u(v.x); o.y = f2bf_u(v.y); o.z = f2bf_u(v.z); o.w = f2bf_u(v.w);
  reinterpret_cast<ushort4*>(out)[i] = o;
}

// ---------------------------------------------------------------------------
// Kernel 2/5: C[M,N] = A[M,K] * B[N,K]^T   (A,B bf16 K-contiguous row-major)
// Workgroup-cooperative 128x64 block tile, 8 waves of 32x32.
// Double-buffered TDM staging: each wave DMAs its 16-row A slice (waves 0-3
// also a 16-row B slice) of stage s+1 while stage s is consumed from LDS.
// Fragment layout per ISA 7.12.2: A lane r(<16): row M=r, K in {0..7,16..23};
// lane r+16: K in {8..15,24..31} -> two 16B loads at klo and klo+16.
// ---------------------------------------------------------------------------
template <bool BF16OUT>
__global__ __launch_bounds__(256) void gemm_tdm_bf16(
    const unsigned short* __restrict__ A, const unsigned short* __restrict__ Bm,
    float* __restrict__ Cf, unsigned short* __restrict__ Cb,
    int M, int N, int K) {
  __shared__ unsigned short Atile[2][128 * 32];   // 16 KiB
  __shared__ unsigned short Btile[2][64 * 32];    //  8 KiB
  const int ln  = threadIdx.x & 31;
  const int wvb = __builtin_amdgcn_readfirstlane((int)threadIdx.x) >> 5; // 0..7
  const int wi  = wvb >> 1;                       // 0..3 : M sub-tile
  const int wj  = wvb & 1;                        // 0..1 : N sub-tile
  const int nb  = N >> 6;
  const int ib  = blockIdx.x / nb;
  const int jb  = blockIdx.x - ib * nb;
  const int r   = ln & 15;
  const int klo = (ln < 16) ? 0 : 8;

  const int aRow0 = ib * 128 + wvb * 16;          // this wave's A slice
  const int bRow0 = jb * 64 + (wvb & 3) * 16;     // waves 0-3: B slice
  const bool doB  = (wvb < 4);

  v8f c00 = {0.f, 0.f, 0.f, 0.f, 0.f, 0.f, 0.f, 0.f};
  v8f c01 = c00, c10 = c00, c11 = c00;

  auto issue = [&](int buf, int k0) {
    tdm_load_tile_bf16(lds_offset(&Atile[buf][wvb * 16 * 32]),
                       A + (size_t)aRow0 * K + k0,
                       (unsigned)K, (unsigned)M, 32u, 16u, (unsigned)K);
    if (doB)
      tdm_load_tile_bf16(lds_offset(&Btile[buf][(wvb & 3) * 16 * 32]),
                         Bm + (size_t)bRow0 * K + k0,
                         (unsigned)K, (unsigned)N, 32u, 16u, (unsigned)K);
  };
  auto compute = [&](int buf) {
    const unsigned short* a0p = &Atile[buf][(wi * 32 + r) * 32];
    const unsigned short* a1p = a0p + 16 * 32;
    const unsigned short* b0p = &Btile[buf][(wj * 32 + r) * 32];
    const unsigned short* b1p = b0p + 16 * 32;
    v16bf a0, a1, b0, b1;
    *((v8bf*)&a0)     = *(const v8bf*)(a0p + klo);
    *((v8bf*)&a0 + 1) = *(const v8bf*)(a0p + klo + 16);
    *((v8bf*)&a1)     = *(const v8bf*)(a1p + klo);
    *((v8bf*)&a1 + 1) = *(const v8bf*)(a1p + klo + 16);
    *((v8bf*)&b0)     = *(const v8bf*)(b0p + klo);
    *((v8bf*)&b0 + 1) = *(const v8bf*)(b0p + klo + 16);
    *((v8bf*)&b1)     = *(const v8bf*)(b1p + klo);
    *((v8bf*)&b1 + 1) = *(const v8bf*)(b1p + klo + 16);
    c00 = WMMA_BF16(a0, b0, c00);
    c01 = WMMA_BF16(a0, b1, c01);
    c10 = WMMA_BF16(a1, b0, c10);
    c11 = WMMA_BF16(a1, b1, c11);
  };

  const int nk = K >> 5;
  issue(0, 0);
  for (int kc = 0; kc < nk - 1; ++kc) {
    issue((kc + 1) & 1, (kc + 1) * 32);            // stage s+1 DMA in flight
    __builtin_amdgcn_s_wait_tensorcnt(1);          // oldest (stage s) landed
    __syncthreads();
    compute(kc & 1);
    __syncthreads();                               // release buf before reuse
  }
  __builtin_amdgcn_s_wait_tensorcnt(0);
  __syncthreads();
  compute((nk - 1) & 1);

  // C/D layout: VGPR j, lane c -> row = j + 8*(c>=16), col = c%16
  const int half8 = (ln < 16) ? 0 : 8;
  const int col   = jb * 64 + wj * 32 + r;
#pragma unroll
  for (int j = 0; j < 8; ++j) {
    const size_t r0 = (size_t)(ib * 128 + wi * 32 + j + half8) * N + col;
    const size_t r1 = r0 + (size_t)16 * N;
    if constexpr (BF16OUT) {
      Cb[r0] = f2bf_u(c00[j]); Cb[r0 + 16] = f2bf_u(c01[j]);
      Cb[r1] = f2bf_u(c10[j]); Cb[r1 + 16] = f2bf_u(c11[j]);
    } else {
      Cf[r0] = c00[j]; Cf[r0 + 16] = c01[j];
      Cf[r1] = c10[j]; Cf[r1 + 16] = c11[j];
    }
  }
}

// ---------------------------------------------------------------------------
// Kernel 3: RoPE(q,k) + head split. Q,K -> [BH,T,64]; V -> transposed [BH,64,T]
// so V rows are K-major for the P@V WMMA B-operand.
// ---------------------------------------------------------------------------
__global__ __launch_bounds__(256) void rope_split(
    const unsigned short* __restrict__ qkv, const float* __restrict__ cosb,
    const float* __restrict__ sinb, unsigned short* __restrict__ Qo,
    unsigned short* __restrict__ Ko, unsigned short* __restrict__ Vo) {
  const int idx = blockIdx.x * 256 + threadIdx.x;
  if (idx >= BH_ * T_) return;
  const int t  = idx & (T_ - 1);
  const int bh = idx >> 11;
  const int b  = bh >> 4;
  const int h  = bh & 15;
  const unsigned short* row = qkv + (size_t)(b * T_ + t) * (3 * DIM_);
  const unsigned short* qr = row + h * HD_;
  const unsigned short* kr = row + DIM_ + h * HD_;
  const unsigned short* vr = row + 2 * DIM_ + h * HD_;
  unsigned short* qo = Qo + ((size_t)bh * T_ + t) * HD_;
  unsigned short* ko = Ko + ((size_t)bh * T_ + t) * HD_;
  unsigned short* vo = Vo + (size_t)bh * HD_ * T_ + t;
#pragma unroll 4
  for (int d = 0; d < 32; ++d) {
    const float c  = cosb[t * 32 + d];
    const float s  = sinb[t * 32 + d];
    const float q1 = bf2f(qr[d]), q2 = bf2f(qr[d + 32]);
    const float k1 = bf2f(kr[d]), k2 = bf2f(kr[d + 32]);
    qo[d]      = f2bf_u(q1 * c - q2 * s);
    qo[d + 32] = f2bf_u(q1 * s + q2 * c);
    ko[d]      = f2bf_u(k1 * c - k2 * s);
    ko[d + 32] = f2bf_u(k1 * s + k2 * c);
    vo[(size_t)d * T_]        = vr[d];
    vo[(size_t)(d + 32) * T_] = vr[d + 32];
  }
}

// ---------------------------------------------------------------------------
// Kernel 4: causal flash attention. One wave32 per (bh, 16-query tile).
// S = Q Kt (two K=32 WMMAs per 16-key tile), fp32 online softmax with
// 16-lane butterfly reductions, P transposed through LDS, O += P V (4 WMMAs).
// Output written bf16 as [B*T, H*64] so the out-proj GEMM consumes it directly.
// ---------------------------------------------------------------------------
__global__ __launch_bounds__(128) void flash_attn(
    const unsigned short* __restrict__ Qm, const unsigned short* __restrict__ Km,
    const unsigned short* __restrict__ Vtm, unsigned short* __restrict__ Om) {
  __shared__ unsigned short Pl[4][16 * 32];   // per-wave 16x32 bf16 P tile
  const int ln    = threadIdx.x & 31;
  const int wvb   = threadIdx.x >> 5;
  const int gw    = (blockIdx.x << 2) | wvb;
  const int bh    = gw >> 7;                  // T/16 = 128 query tiles per head
  const int qt    = gw & 127;
  const int q0    = qt << 4;
  const int r     = ln & 15;
  const int klo   = (ln < 16) ? 0 : 8;
  const int half8 = (ln < 16) ? 0 : 8;

  // Q fragments (A-operand), resident for the whole key loop
  const bf16_t* qrow = (const bf16_t*)Qm + ((size_t)bh * T_ + q0 + r) * HD_;
  v16bf aq0, aq1;
  *((v8bf*)&aq0)     = *(const v8bf*)(qrow + klo);
  *((v8bf*)&aq0 + 1) = *(const v8bf*)(qrow + klo + 16);
  *((v8bf*)&aq1)     = *(const v8bf*)(qrow + 32 + klo);
  *((v8bf*)&aq1 + 1) = *(const v8bf*)(qrow + 32 + klo + 16);

  v8f zero = {0.f, 0.f, 0.f, 0.f, 0.f, 0.f, 0.f, 0.f};
  v8f o0 = zero, o1 = zero, o2 = zero, o3 = zero;
  float m[8], l[8];
#pragma unroll
  for (int j = 0; j < 8; ++j) { m[j] = -__builtin_inff(); l[j] = 0.f; }

  const float scale = 0.125f;                 // 1/sqrt(64)
  const int nk = (q0 + 16 + 31) >> 5;         // 32-key chunks, causal bound
  for (int kc = 0; kc < nk; ++kc) {
    const int kb = kc << 5;
    const bf16_t* k0p = (const bf16_t*)Km + ((size_t)bh * T_ + kb + r) * HD_;
    const bf16_t* k1p = k0p + 16 * HD_;
    v16bf bk;
    v8f s0 = zero, s1 = zero;
    *((v8bf*)&bk)     = *(const v8bf*)(k0p + klo);
    *((v8bf*)&bk + 1) = *(const v8bf*)(k0p + klo + 16);
    s0 = WMMA_BF16(aq0, bk, s0);
    *((v8bf*)&bk)     = *(const v8bf*)(k0p + 32 + klo);
    *((v8bf*)&bk + 1) = *(const v8bf*)(k0p + 32 + klo + 16);
    s0 = WMMA_BF16(aq1, bk, s0);
    *((v8bf*)&bk)     = *(const v8bf*)(k1p + klo);
    *((v8bf*)&bk + 1) = *(const v8bf*)(k1p + klo + 16);
    s1 = WMMA_BF16(aq0, bk, s1);
    *((v8bf*)&bk)     = *(const v8bf*)(k1p + 32 + klo);
    *((v8bf*)&bk + 1) = *(const v8bf*)(k1p + 32 + klo + 16);
    s1 = WMMA_BF16(aq1, bk, s1);

    // causal mask + scale + row max (rows live in 16-lane halves)
    float mn[8];
#pragma unroll
    for (int j = 0; j < 8; ++j) {
      const int qrowi = q0 + j + half8;
      float v0 = (kb + r <= qrowi)      ? s0[j] * scale : -__builtin_inff();
      float v1 = (kb + 16 + r <= qrowi) ? s1[j] * scale : -__builtin_inff();
      s0[j] = v0; s1[j] = v1;
      float rm = fmaxf(v0, v1);
      rm = fmaxf(rm, __shfl_xor(rm, 1, 32));
      rm = fmaxf(rm, __shfl_xor(rm, 2, 32));
      rm = fmaxf(rm, __shfl_xor(rm, 4, 32));
      rm = fmaxf(rm, __shfl_xor(rm, 8, 32));
      mn[j] = fmaxf(m[j], rm);
    }
    // exp, row sums, rescale accumulators, stage P (bf16) into LDS
#pragma unroll
    for (int j = 0; j < 8; ++j) {
      const float p0 = __expf(s0[j] - mn[j]);
      const float p1 = __expf(s1[j] - mn[j]);
      float rs = p0 + p1;
      rs += __shfl_xor(rs, 1, 32);
      rs += __shfl_xor(rs, 2, 32);
      rs += __shfl_xor(rs, 4, 32);
      rs += __shfl_xor(rs, 8, 32);
      const float al = __expf(m[j] - mn[j]);
      l[j] = l[j] * al + rs;
      m[j] = mn[j];
      o0[j] *= al; o1[j] *= al; o2[j] *= al; o3[j] *= al;
      const int row = j + half8;
      Pl[wvb][row * 32 + r]      = f2bf_u(p0);
      Pl[wvb][row * 32 + 16 + r] = f2bf_u(p1);
    }
    // same-wave LDS ops are in-order; wait + compiler barrier before re-read
    __asm__ volatile("s_wait_dscnt 0" ::: "memory");

    // reload P in A-operand layout (transpose through LDS)
    v16bf pa;
    const unsigned short* plp = &Pl[wvb][r * 32];
    *((v8bf*)&pa)     = *(const v8bf*)(plp + klo);
    *((v8bf*)&pa + 1) = *(const v8bf*)(plp + klo + 16);

    // O += P @ V : Vt rows are d-indexed, key-contiguous (B-operand)
    const bf16_t* vbp = (const bf16_t*)Vtm + ((size_t)bh * HD_ + r) * T_ + kb;
    v16bf bv;
    *((v8bf*)&bv)     = *(const v8bf*)(vbp + klo);
    *((v8bf*)&bv + 1) = *(const v8bf*)(vbp + klo + 16);
    o0 = WMMA_BF16(pa, bv, o0);
    vbp += (size_t)16 * T_;
    *((v8bf*)&bv)     = *(const v8bf*)(vbp + klo);
    *((v8bf*)&bv + 1) = *(const v8bf*)(vbp + klo + 16);
    o1 = WMMA_BF16(pa, bv, o1);
    vbp += (size_t)16 * T_;
    *((v8bf*)&bv)     = *(const v8bf*)(vbp + klo);
    *((v8bf*)&bv + 1) = *(const v8bf*)(vbp + klo + 16);
    o2 = WMMA_BF16(pa, bv, o2);
    vbp += (size_t)16 * T_;
    *((v8bf*)&bv)     = *(const v8bf*)(vbp + klo);
    *((v8bf*)&bv + 1) = *(const v8bf*)(vbp + klo + 16);
    o3 = WMMA_BF16(pa, bv, o3);
  }

  // epilogue: normalize and store bf16 into [B*T, H*64]
  const int b = bh >> 4, h = bh & 15;
#pragma unroll
  for (int j = 0; j < 8; ++j) {
    const float inv = 1.0f / l[j];
    const int   t   = q0 + j + half8;
    const size_t ro = (size_t)(b * T_ + t) * DIM_ + h * HD_ + r;
    Om[ro]      = f2bf_u(o0[j] * inv);
    Om[ro + 16] = f2bf_u(o1[j] * inv);
    Om[ro + 32] = f2bf_u(o2[j] * inv);
    Om[ro + 48] = f2bf_u(o3[j] * inv);
  }
}

// ---------------------------------------------------------------------------
extern "C" void kernel_launch(void* const* d_in, const int* in_sizes, int n_in,
                              void* d_out, int out_size, void* d_ws, size_t ws_size,
                              hipStream_t stream) {
  (void)in_sizes; (void)n_in; (void)out_size; (void)ws_size;
  const float* x    = (const float*)d_in[0];
  const float* cosb = (const float*)d_in[1];
  const float* sinb = (const float*)d_in[2];
  const float* wqkv = (const float*)d_in[3];
  const float* wout = (const float*)d_in[4];
  float* out = (float*)d_out;
  char*  ws  = (char*)d_ws;

  // workspace layout (bytes)
  unsigned short* Xb    = (unsigned short*)(ws + 0);          //  8 MiB: x bf16 [4096,1024]
  unsigned short* Wqkvb = (unsigned short*)(ws + 8388608);    //  6 MiB: w_qkv bf16 [3072,1024]
  unsigned short* Woutb = (unsigned short*)(ws + 14680064);   //  2 MiB: w_out bf16 [1024,1024]
  unsigned short* QKVb  = (unsigned short*)(ws + 16777216);   // 24 MiB: qkv bf16 [4096,3072]
  unsigned short* Qb    = (unsigned short*)(ws + 41943040);   //  8 MiB: [BH,T,64]
  unsigned short* Kb    = (unsigned short*)(ws + 50331648);   //  8 MiB: [BH,T,64]
  unsigned short* Vt    = (unsigned short*)(ws + 58720256);   //  8 MiB: [BH,64,T]
  unsigned short* Ob    = (unsigned short*)(ws + 67108864);   //  8 MiB: attn out bf16 [4096,1024]

  cast_f32_to_bf16<<<4096, 256, 0, stream>>>(x,    Xb,    1048576); // 4096*1024/4
  cast_f32_to_bf16<<<3072, 256, 0, stream>>>(wqkv, Wqkvb, 786432);  // 3072*1024/4
  cast_f32_to_bf16<<<1024, 256, 0, stream>>>(wout, Woutb, 262144);  // 1024*1024/4

  // QKV projection: [4096,3072] = Xb @ Wqkvb^T ; (4096/128)*(3072/64) = 1536 blocks
  gemm_tdm_bf16<true><<<1536, 256, 0, stream>>>(Xb, Wqkvb, nullptr, QKVb, 4096, 3072, 1024);

  rope_split<<<256, 256, 0, stream>>>(QKVb, cosb, sinb, Qb, Kb, Vt);

  // 32 heads * 128 query tiles = 4096 waves, 4 waves/block
  flash_attn<<<1024, 128, 0, stream>>>(Qb, Kb, Vt, Ob);

  // out projection: [4096,1024] = Ob @ Woutb^T ; (4096/128)*(1024/64) = 512 blocks
  gemm_tdm_bf16<false><<<512, 256, 0, stream>>>(Ob, Woutb, out, nullptr, 4096, 1024, 1024);
}
